// GraphAttentionLayer11_60662118088926
// MI455X (gfx1250) — matrix operand order repaired and checked by reading, not compiled, not run on hardware
//
#include <hip/hip_runtime.h>
#include <hip/hip_bf16.h>

typedef __attribute__((ext_vector_type(16))) _Float16 v16h;
typedef __attribute__((ext_vector_type(8)))  _Float16 v8h;
typedef __attribute__((ext_vector_type(4)))  _Float16 v4h;
typedef __attribute__((ext_vector_type(8)))  float    v8f;

// async-to-LDS builtin operates on v4i32 pointers in AS1 (global) / AS3 (LDS)
typedef __attribute__((ext_vector_type(4))) int gat_i4;
typedef __attribute__((address_space(1))) gat_i4 gat_i4_as1;
typedef __attribute__((address_space(3))) gat_i4 gat_i4_as3;

#define B_ 4
#define N_ 2048
#define F_ 256
#define H_ 8
#define O_ 64
#define LRELU_ALPHA 0.2f
#define NEG_HUGE -3.0e38f

// gfx1250 async global->LDS copy (ASYNCcnt); fall back to plain copy if the
// toolchain doesn't expose the builtins.
#if __has_builtin(__builtin_amdgcn_global_load_async_to_lds_b128) && \
    __has_builtin(__builtin_amdgcn_s_wait_asynccnt)
#define GAT_HAS_ASYNC 1
#else
#define GAT_HAS_ASYNC 0
#endif

__device__ __forceinline__ void gat_async_copy_b128(const void* g, void* l) {
#if GAT_HAS_ASYNC
  __builtin_amdgcn_global_load_async_to_lds_b128(
      (gat_i4_as1*)(unsigned long long)g,
      (gat_i4_as3*)(unsigned int)(unsigned long long)l, 0, 0);
#else
  *(float4*)l = *(const float4*)g;
#endif
}
__device__ __forceinline__ void gat_async_wait() {
#if GAT_HAS_ASYNC
  __builtin_amdgcn_s_wait_asynccnt(0);
#endif
}

// ---------------------------------------------------------------------------
// h (B,N,F) f32 -> f16 (same layout)
// ---------------------------------------------------------------------------
__global__ void gat_conv_h(const float* __restrict__ src,
                           _Float16* __restrict__ dst, int n) {
  int i = blockIdx.x * blockDim.x + threadIdx.x;
  if (i < n) dst[i] = (_Float16)src[i];
}

// ---------------------------------------------------------------------------
// attn_weights (H,F,O) f32 -> f16 transposed to (H,O,F) so the WMMA B
// fragment is two contiguous 16B loads.
// ---------------------------------------------------------------------------
__global__ void gat_conv_wT(const float* __restrict__ w,
                            _Float16* __restrict__ wT) {
  int idx = blockIdx.x * blockDim.x + threadIdx.x;
  if (idx >= H_ * F_ * O_) return;
  int o = idx % O_;
  int f = (idx / O_) % F_;
  int h = idx / (F_ * O_);
  wT[((size_t)h * O_ + o) * F_ + f] = (_Float16)w[idx];
}

// ---------------------------------------------------------------------------
// waj[h,f] = sum_o W[h,f,o]*a_j[h,o], wai likewise (tiny, f32).
// ---------------------------------------------------------------------------
__global__ void gat_waj(const float* __restrict__ w,
                        const float* __restrict__ coeffs,
                        float* __restrict__ waj, float* __restrict__ wai) {
  int idx = blockIdx.x * blockDim.x + threadIdx.x;  // H*F
  if (idx >= H_ * F_) return;
  int h = idx / F_;
  const float* wrow = w + (size_t)idx * O_;
  const float* aj = coeffs + h * 2 * O_;
  const float* ai = aj + O_;
  float sj = 0.f, si = 0.f;
#pragma unroll
  for (int o = 0; o < O_; ++o) {
    float x = wrow[o];
    sj += x * aj[o];
    si += x * ai[o];
  }
  waj[idx] = sj;
  wai[idx] = si;
}

// ---------------------------------------------------------------------------
// s[b,h,n] = h[b,n,:].waj[h,:], d likewise with wai (full f32 precision).
// ---------------------------------------------------------------------------
__global__ void gat_sd(const float* __restrict__ hsrc,
                       const float* __restrict__ waj,
                       const float* __restrict__ wai,
                       float* __restrict__ s, float* __restrict__ d) {
  int tid = blockIdx.x * blockDim.x + threadIdx.x;  // B*H*N
  if (tid >= B_ * H_ * N_) return;
  int n = tid % N_;
  int h = (tid / N_) % H_;
  int b = tid / (N_ * H_);
  const float4* hrow = (const float4*)(hsrc + ((size_t)b * N_ + n) * F_);
  const float4* aj4 = (const float4*)(waj + h * F_);
  const float4* ai4 = (const float4*)(wai + h * F_);
  float ss = 0.f, dd = 0.f;
#pragma unroll 8
  for (int q = 0; q < F_ / 4; ++q) {
    float4 x = hrow[q], a = aj4[q], c = ai4[q];
    ss += x.x * a.x + x.y * a.y + x.z * a.z + x.w * a.w;
    dd += x.x * c.x + x.y * c.y + x.z * c.z + x.w * c.w;
  }
  s[tid] = ss;
  d[tid] = dd;
}

// ---------------------------------------------------------------------------
// whT[b,h,o,n] = (h[b] @ W[h])^T in f16. Block = 128 threads / 4 waves,
// each wave one 16x16 tile over K=256 (8 x v_wmma_f32_16x16x32_f16).
// A and B fragments are contiguous v8h global loads; D stored as one packed
// b128 per lane (column-major output layout makes lane rows consecutive).
// ---------------------------------------------------------------------------
__global__ __launch_bounds__(128) void gat_wh_gemm(
    const _Float16* __restrict__ hf, const _Float16* __restrict__ wT,
    _Float16* __restrict__ whT) {
  int t = threadIdx.x;
  int tile = blockIdx.x;
  int nt = tile & 127;        // N/16 row tiles
  int h  = (tile >> 7) & 7;
  int b  = tile >> 10;
  int i0 = nt * 16;

  int lane = t & 31, wave = t >> 5;
  int hi = lane >> 4;         // 0: lanes 0-15, 1: lanes 16-31
  int col = lane & 15;
  int row = lane & 15;
  int o0 = wave * 16;

  const _Float16* arow = hf + ((size_t)(b * N_) + i0 + row) * F_;
  const _Float16* bcol = wT + ((size_t)h * O_ + o0 + col) * F_;

  v8f acc = {};
  for (int k0 = 0; k0 < F_; k0 += 32) {
    // A 16x32 f16: elems 0-7 = K k0+8*hi..+7 ; elems 8-15 = +16 more
    union { v16h v; v8h g[2]; } a;
    int g0 = k0 + 8 * hi;
    a.g[0] = *(const v8h*)(arow + g0);
    a.g[1] = *(const v8h*)(arow + g0 + 16);
    // B 32x16 f16: lane col, elems 0-7 = K k0+8*hi..+7 ; elems 8-15 = +16 more
    union { v16h v; v8h g[2]; } bb;
    const _Float16* bp = bcol + k0 + 8 * hi;
    bb.g[0] = *(const v8h*)(bp);
    bb.g[1] = *(const v8h*)(bp + 16);
    acc = __builtin_amdgcn_wmma_f32_16x16x32_f16(false, a.v, false, bb.v,
                                                 (short)0, acc, false, false);
  }
  // D: VGPR v -> (M = v + 8*hi, col). In whT layout the 8 M's are consecutive.
  union { v8h v; _Float16 e[8]; } st;
#pragma unroll
  for (int v = 0; v < 8; ++v) st.e[v] = (_Float16)acc[v];
  *(v8h*)(whT + (((size_t)(b * H_ + h)) * O_ + o0 + col) * N_ + i0 + 8 * hi) = st.v;
}

// ---------------------------------------------------------------------------
// Fused masked softmax + aggregation. One block per (b, 16-row tile).
// adj packed to a 4KB LDS bitmask once (adj hits HBM exactly once total).
// Pass 1 is a masked max of s (lrelu hoisted out by monotonicity);
// softmax denominator is accumulated during P-tile generation in pass 2.
// A fragment: 2 x b128 LDS reads; B fragment: 2 x b128 global (L2) reads.
// ---------------------------------------------------------------------------
__global__ __launch_bounds__(128) void gat_attn(
    const int* __restrict__ adj, const _Float16* __restrict__ whT,
    const float* __restrict__ sv, const float* __restrict__ dv,
    const float* __restrict__ bias, float* __restrict__ out) {
  __shared__ __align__(16) float s_s[N_];          // 8KB
  __shared__ unsigned s_adj[16 * 64];              // 4KB bitmask
  __shared__ float s_d[16], s_m[16], s_linv[16];
  __shared__ float s_red[16][8];
  __shared__ __align__(16) _Float16 s_p[2][16 * 32];  // double-buffered P

  int t = threadIdx.x;
  int b = blockIdx.x >> 7;
  int i0 = (blockIdx.x & 127) * 16;

  // ---- pack adjacency rows to bits ----
#pragma unroll
  for (int q = 0; q < 8; ++q) {
    int w = t * 8 + q;                  // 1024 words
    int r = w >> 6, jw = w & 63;
    const int4* p = (const int4*)(adj + ((size_t)b * N_ + i0 + r) * N_ + jw * 32);
    unsigned bits = 0;
#pragma unroll
    for (int v4 = 0; v4 < 8; ++v4) {
      int4 x = p[v4];
      bits |= (unsigned)(x.x != 0) << (v4 * 4 + 0);
      bits |= (unsigned)(x.y != 0) << (v4 * 4 + 1);
      bits |= (unsigned)(x.z != 0) << (v4 * 4 + 2);
      bits |= (unsigned)(x.w != 0) << (v4 * 4 + 3);
    }
    s_adj[w] = bits;
  }

  int lane = t & 31, wave = t >> 5;
  int hi = lane >> 4;
  int col = lane & 15;
  int o0 = wave * 16;
  int r = t & 15, g = t >> 4;           // pass-1 / P-tile mapping

  for (int h = 0; h < H_; ++h) {
    __syncthreads();
    // ---- stage s[j] row (8KB) via async-to-LDS, + d for our 16 rows ----
    {
      const float* src = sv + ((size_t)b * H_ + h) * N_;
#pragma unroll
      for (int q = 0; q < 4; ++q) {
        int idx = (t + 128 * q) * 4;
        gat_async_copy_b128(src + idx, &s_s[idx]);
      }
      gat_async_wait();
    }
    if (t < 16) s_d[t] = dv[((size_t)b * H_ + h) * N_ + i0 + t];
    __syncthreads();

    // ---- pass 1: masked max of s_j per row; m = lrelu(d + max) ----
    float mx = NEG_HUGE;
    const float4* sv4 = (const float4*)s_s;
    for (int q = 0; q < 8; ++q) {
      unsigned bits = s_adj[r * 64 + g * 8 + q];
      int base4 = g * 64 + q * 8;
#pragma unroll
      for (int w4 = 0; w4 < 8; ++w4) {
        float4 vv = sv4[base4 + w4];
        unsigned bb = bits >> (w4 * 4);
        if (bb & 1u) mx = fmaxf(mx, vv.x);
        if (bb & 2u) mx = fmaxf(mx, vv.y);
        if (bb & 4u) mx = fmaxf(mx, vv.z);
        if (bb & 8u) mx = fmaxf(mx, vv.w);
      }
    }
    s_red[r][g] = mx;
    __syncthreads();
    if (t < 16) {
      float m = s_red[t][0];
#pragma unroll
      for (int q = 1; q < 8; ++q) m = fmaxf(m, s_red[t][q]);
      float e = s_d[t] + m;             // lrelu is monotone increasing
      s_m[t] = e >= 0.f ? e : LRELU_ALPHA * e;
    }
    __syncthreads();

    // ---- pass 2: stream j-tiles, accumulate l alongside P, WMMA ----
    v8f acc = {};
    float l_part = 0.f;
    float d_r = s_d[r], m_r = s_m[r];
    const _Float16* bbase = whT + ((size_t)(b * H_ + h) * O_ + o0 + col) * N_;
    for (int jt = 0; jt < N_ / 32; ++jt) {
      int j0 = jt * 32;
      int par = jt & 1;
      // P tile: thread owns row r, cols g*4..g*4+3; one packed b64 store
      union { v4h v; _Float16 e[4]; } pk;
#pragma unroll
      for (int q = 0; q < 4; ++q) {
        int j = j0 + g * 4 + q;
        float e = d_r + s_s[j];
        e = e >= 0.f ? e : LRELU_ALPHA * e;
        unsigned bit = (s_adj[r * 64 + (j >> 5)] >> (j & 31)) & 1u;
        float p = bit ? __expf(e - m_r) : 0.f;
        l_part += p;
        pk.e[q] = (_Float16)p;
      }
      *(v4h*)&s_p[par][r * 32 + g * 4] = pk.v;
      __syncthreads();
      // A fragment from P (two contiguous b128 LDS reads)
      union { v16h v; v8h g2[2]; } af;
      af.g2[0] = *(const v8h*)&s_p[par][col * 32 + 8 * hi];
      af.g2[1] = *(const v8h*)&s_p[par][col * 32 + 8 * hi + 16];
      // B fragment from whT (two contiguous b128 global reads, L2-resident)
      union { v16h v; v8h g2[2]; } bf;
      const _Float16* bp = bbase + j0 + 8 * hi;
      bf.g2[0] = *(const v8h*)(bp);
      bf.g2[1] = *(const v8h*)(bp + 16);
      acc = __builtin_amdgcn_wmma_f32_16x16x32_f16(false, af.v, false, bf.v,
                                                   (short)0, acc, false, false);
    }

    // ---- reduce softmax denominator ----
    s_red[r][g] = l_part;
    __syncthreads();
    if (t < 16) {
      float l = 0.f;
#pragma unroll
      for (int q = 0; q < 8; ++q) l += s_red[t][q];
      s_linv[t] = l > 0.f ? 1.f / l : 0.f;
    }
    __syncthreads();

    // ---- epilogue ----
#pragma unroll
    for (int v = 0; v < 8; ++v) {
      int M = v + 8 * hi;
      float o = acc[v] * s_linv[M] + bias[h * O_ + o0 + col];
      out[((size_t)b * N_ + i0 + M) * (H_ * O_) + h * O_ + o0 + col] = o;
    }
  }
}

// ---------------------------------------------------------------------------
extern "C" void kernel_launch(void* const* d_in, const int* in_sizes, int n_in,
                              void* d_out, int out_size, void* d_ws, size_t ws_size,
                              hipStream_t stream) {
  const float* h_f32  = (const float*)d_in[0];   // (B,N,F)
  const int*   adj    = (const int*)  d_in[1];   // (B,N,N)
  const float* w_f32  = (const float*)d_in[2];   // (H,F,O)
  const float* bias   = (const float*)d_in[3];   // (H,O)
  const float* coeffs = (const float*)d_in[4];   // (H,2O,1)
  float* out = (float*)d_out;

  char* ws = (char*)d_ws;
  _Float16* hf16 = (_Float16*)(ws);               // 4,194,304 B
  _Float16* w16T = (_Float16*)(ws + 4194304);     //   262,144 B  (H,O,F)
  _Float16* whT  = (_Float16*)(ws + 4456448);     // 8,388,608 B  (B,H,O,N)
  float*    sbuf = (float*)   (ws + 12845056);    //   262,144 B
  float*    dbuf = (float*)   (ws + 13107200);    //   262,144 B
  float*    waj  = (float*)   (ws + 13369344);    //     8,192 B
  float*    wai  = (float*)   (ws + 13377536);    //     8,192 B

  const int n_h = B_ * N_ * F_;   // 2,097,152
  const int n_w = H_ * F_ * O_;   //   131,072
  gat_conv_h<<<(n_h + 255) / 256, 256, 0, stream>>>(h_f32, hf16, n_h);
  gat_conv_wT<<<(n_w + 255) / 256, 256, 0, stream>>>(w_f32, w16T);
  gat_waj<<<(H_ * F_ + 255) / 256, 256, 0, stream>>>(w_f32, coeffs, waj, wai);

  gat_wh_gemm<<<B_ * H_ * (N_ / 16), 128, 0, stream>>>(hf16, w16T, whT);

  const int n_sd = B_ * H_ * N_;  // 65,536
  gat_sd<<<(n_sd + 255) / 256, 256, 0, stream>>>(h_f32, waj, wai, sbuf, dbuf);

  gat_attn<<<B_ * (N_ / 16), 128, 0, stream>>>(adj, whT, sbuf, dbuf, bias, out);
}
